// BLA_30305289241008
// MI455X (gfx1250) — matrix operand order repaired and checked by reading, not compiled, not run on hardware
//
#include <hip/hip_runtime.h>

// ---------------------------------------------------------------------------
// Types for CDNA5 WMMA
// ---------------------------------------------------------------------------
typedef __attribute__((ext_vector_type(16))) __bf16 v16bf;
typedef __attribute__((ext_vector_type(8)))  float  v8f;

union Frag { uint4 u[2]; v16bf v; };

#if defined(__has_builtin)
#if __has_builtin(__builtin_amdgcn_cvt_pk_bf16_f32)
#define HAVE_CVT_PK_BF16 1
#endif
#endif

__device__ __forceinline__ unsigned int pk2bf(float a, float b) {
#ifdef HAVE_CVT_PK_BF16
    auto t = __builtin_amdgcn_cvt_pk_bf16_f32(a, b);   // v_cvt_pk_bf16_f32 (RNE)
    static_assert(sizeof(t) == 4, "cvt_pk_bf16 size");
    unsigned int r;
    __builtin_memcpy(&r, &t, 4);
    return r;
#else
    // fp32 -> bf16 round-half-up, packed pair: 2x v_add + 1x v_perm_b32
    const unsigned int ua = __float_as_uint(a) + 0x8000u;
    const unsigned int ub = __float_as_uint(b) + 0x8000u;
    // perm(hi=ub, lo=ua): bytes {2,3} of ua (its high half) -> result[15:0],
    //                     bytes {6,7} = ub high half        -> result[31:16]
    return __builtin_amdgcn_perm(ub, ua, 0x07060302u);
#endif
}

// fp32 -> bf16 bulk conversion (4 elements / thread)
__global__ void k_cvtbf(const float* __restrict__ in, unsigned short* __restrict__ out,
                        int total4)
{
    const int idx = blockIdx.x * 256 + threadIdx.x;
    if (idx >= total4) return;
    float4 v = ((const float4*)in)[idx];
    uint2 p;
    p.x = pk2bf(v.x, v.y);
    p.y = pk2bf(v.z, v.w);
    ((uint2*)out)[idx] = p;
}

// ---------------------------------------------------------------------------
// GEMM: C[M,N] = A[M,K] @ W[N,K]^T (+ bias[N]) (+ ReLU)
//   A fp32 (converted to bf16 while staging into LDS)
//   W bf16 (pre-converted; staged with GLOBAL_LOAD_ASYNC_TO_LDS_B128)
// Block tile 128x64, BK=32, 256 threads = 8 waves (4 in M x 2 in N),
// each wave owns a 32x32 C tile = 4 v_wmma_f32_16x16x32_bf16 accumulators.
// Double-buffered LDS: one barrier per k-step.
// Requires M%128==0, N%64==0, K%32==0 (true for all shapes here).
// ---------------------------------------------------------------------------
template <bool RELU>
__global__ void __launch_bounds__(256)
k_gemm(const float* __restrict__ A, const unsigned short* __restrict__ Wb,
       const float* __restrict__ bias, float* __restrict__ C,
       int M, int N, int K)
{
    __shared__ __align__(16) unsigned short lA[2][128 * 32];
    __shared__ __align__(16) unsigned short lB[2][64 * 32];

    const int tid   = threadIdx.x;
    const int mBase = blockIdx.y * 128;
    const int nBase = blockIdx.x * 64;

    // A staging: thread -> (row 0..127, 16-float half-row)
    const int rowA = tid >> 1;
    const int segA = (tid & 1) * 16;
    const float* gA = A + (size_t)(mBase + rowA) * K + segA;

    // B staging (async, bf16): thread -> (row 0..63, 8-elem segment) = 16 bytes
    const int rowB = tid >> 2;
    const int segB = (tid & 3) * 8;
    const unsigned short* gB = Wb + (size_t)(nBase + rowB) * K + segB;

    const int wid  = tid >> 5;
    const int lane = tid & 31;
    const int wm   = wid & 3;      // wave M group (0..3) -> 32 rows
    const int wn   = wid >> 2;     // wave N group (0..1) -> 32 cols
    const int l16  = lane & 15;
    const int hi   = lane >> 4;

    v8f acc00 = {0.f,0.f,0.f,0.f,0.f,0.f,0.f,0.f};
    v8f acc01 = acc00, acc10 = acc00, acc11 = acc00;

    float4 ar[4];
    auto loadA = [&](int kt) {
        const float4* p = (const float4*)(gA + kt);
        ar[0] = p[0]; ar[1] = p[1]; ar[2] = p[2]; ar[3] = p[3];
    };
    auto storeA = [&](int buf) {
        unsigned short* s = &lA[buf][rowA * 32 + segA];
        ((uint4*)s)[0] = make_uint4(pk2bf(ar[0].x, ar[0].y), pk2bf(ar[0].z, ar[0].w),
                                    pk2bf(ar[1].x, ar[1].y), pk2bf(ar[1].z, ar[1].w));
        ((uint4*)s)[1] = make_uint4(pk2bf(ar[2].x, ar[2].y), pk2bf(ar[2].z, ar[2].w),
                                    pk2bf(ar[3].x, ar[3].y), pk2bf(ar[3].z, ar[3].w));
    };
    auto issueB = [&](int buf, int kt) {
        // per-lane 16B copy, GV mode (saddr = off), tracked by ASYNCcnt
        unsigned ldsa = (unsigned)(size_t)&lB[buf][rowB * 32 + segB];
        const unsigned short* ga = gB + kt;
        asm volatile("global_load_async_to_lds_b128 %0, %1, off"
                     :: "v"(ldsa), "v"(ga) : "memory");
    };

    // prologue: fill buffer 0
    issueB(0, 0);
    loadA(0);
    storeA(0);
    asm volatile("s_wait_asynccnt 0x0" ::: "memory");
    __syncthreads();

    int cur = 0;
    for (int kt = 0; kt < K; kt += 32) {
        const bool more = (kt + 32 < K);
        if (more) {
            issueB(cur ^ 1, kt + 32);   // async B copy overlaps compute below
            loadA(kt + 32);
            if (kt + 64 < K) __builtin_prefetch(gA + kt + 64, 0, 1); // global_prefetch_b8
        }

        // Fragment loads (per ISA 16-bit VGPR layouts):
        // A 16x32: lane<16 holds K{0..7,16..23}, lane>=16 holds K{8..15,24..31}
        const unsigned short* fa0 = &lA[cur][(wm * 32 +      l16) * 32 + hi * 8];
        const unsigned short* fa1 = &lA[cur][(wm * 32 + 16 + l16) * 32 + hi * 8];
        // B 32x16: lane<16 holds K{0..7,16..23} of col l16, lane>=16 holds K{8..15,24..31}
        const unsigned short* fb0 = &lB[cur][(wn * 32 +      l16) * 32 + hi * 8];
        const unsigned short* fb1 = &lB[cur][(wn * 32 + 16 + l16) * 32 + hi * 8];

        Frag a0, a1, b0, b1;
        a0.u[0] = *(const uint4*)(fa0);   a0.u[1] = *(const uint4*)(fa0 + 16);
        a1.u[0] = *(const uint4*)(fa1);   a1.u[1] = *(const uint4*)(fa1 + 16);
        b0.u[0] = *(const uint4*)(fb0);   b0.u[1] = *(const uint4*)(fb0 + 16);
        b1.u[0] = *(const uint4*)(fb1);   b1.u[1] = *(const uint4*)(fb1 + 16);

        acc00 = __builtin_amdgcn_wmma_f32_16x16x32_bf16(false, a0.v, false, b0.v,
                                                        (short)0, acc00, false, false);
        acc01 = __builtin_amdgcn_wmma_f32_16x16x32_bf16(false, a0.v, false, b1.v,
                                                        (short)0, acc01, false, false);
        acc10 = __builtin_amdgcn_wmma_f32_16x16x32_bf16(false, a1.v, false, b0.v,
                                                        (short)0, acc10, false, false);
        acc11 = __builtin_amdgcn_wmma_f32_16x16x32_bf16(false, a1.v, false, b1.v,
                                                        (short)0, acc11, false, false);

        if (more) {
            storeA(cur ^ 1);
            asm volatile("s_wait_asynccnt 0x0" ::: "memory");
        }
        __syncthreads();
        cur ^= 1;
    }

    // C/D layout: lane<16 -> N=lane, M=r ; lane>=16 -> N=lane-16, M=r+8
    const int col0 = nBase + wn * 32 + l16;
    const int row0 = mBase + wm * 32 + hi * 8;
    const float bv0 = bias ? bias[col0]      : 0.f;
    const float bv1 = bias ? bias[col0 + 16] : 0.f;
#pragma unroll
    for (int r = 0; r < 8; ++r) {
        float v00 = acc00[r] + bv0, v01 = acc01[r] + bv1;
        float v10 = acc10[r] + bv0, v11 = acc11[r] + bv1;
        if (RELU) {
            v00 = fmaxf(v00, 0.f); v01 = fmaxf(v01, 0.f);
            v10 = fmaxf(v10, 0.f); v11 = fmaxf(v11, 0.f);
        }
        C[(size_t)(row0 + r)      * N + col0]      = v00;
        C[(size_t)(row0 + r)      * N + col0 + 16] = v01;
        C[(size_t)(row0 + 16 + r) * N + col0]      = v10;
        C[(size_t)(row0 + 16 + r) * N + col0 + 16] = v11;
    }
}

// ---------------------------------------------------------------------------
// Local 3x3 window attention (self / top-down / bottom-up).
// One wave per (token, head): 8 heads * 32 dims, lane = head-dim channel.
// mode: 0 = same grid, 1 = td (kv at h>>1,w>>1 on coarse), 2 = bu (kv at 2h,2w on fine)
// Writes cat chunk = attn_out + ob_chunk (ob bias folded here).
// ---------------------------------------------------------------------------
__global__ void __launch_bounds__(256)
k_attn(const float* __restrict__ pQ, int sQ, int oQ,
       const float* __restrict__ pKV, int sKV, int oK, int oV,
       const float* __restrict__ pos, const float* __restrict__ bias,
       float* __restrict__ cat, int sC, int oC,
       int H, int W, int Hk, int Wk, int mode)
{
    const float sc = 0.4204482076268573f;   // (256/8)^(-0.25)
    const int token = blockIdx.x;           // b*H*W + h*W + w
    const int w  = token % W;
    const int t2 = token / W;
    const int h  = t2 % H;
    const int b  = t2 / H;
    const int head = threadIdx.x >> 5;
    const int lane = threadIdx.x & 31;
    const int c = head * 32 + lane;

    const float qs = sc * pQ[(size_t)token * sQ + oQ + c];
    const int kyc = (mode == 1) ? (h >> 1) : ((mode == 2) ? (h << 1) : h);
    const int kxc = (mode == 1) ? (w >> 1) : ((mode == 2) ? (w << 1) : w);

    float vv[9], lg[9];
#pragma unroll
    for (int t = 0; t < 9; ++t) {
        const int dy = t / 3 - 1, dx = t % 3 - 1;
        int ky = kyc + dy; ky = ky < 0 ? 0 : (ky >= Hk ? Hk - 1 : ky);
        int kx = kxc + dx; kx = kx < 0 ? 0 : (kx >= Wk ? Wk - 1 : kx);
        const size_t basek = ((size_t)(b * Hk + ky) * Wk + kx) * sKV;
        const float kk = sc * pKV[basek + oK + c] + pos[c * 9 + t];
        vv[t] = pKV[basek + oV + c];
        float p = qs * kk;
#pragma unroll
        for (int o = 16; o > 0; o >>= 1) p += __shfl_xor(p, o, 32);
        lg[t] = p;   // full head dot-product, replicated across lanes
    }
    float m = lg[0];
#pragma unroll
    for (int t = 1; t < 9; ++t) m = fmaxf(m, lg[t]);
    float s = 0.f;
#pragma unroll
    for (int t = 0; t < 9; ++t) { lg[t] = __expf(lg[t] - m); s += lg[t]; }
    float out = 0.f;
#pragma unroll
    for (int t = 0; t < 9; ++t) out += lg[t] * vv[t];
    out *= (1.f / s);

    cat[(size_t)token * sC + oC + c] = out + bias[c];
}

// Fill absent td/bu chunk with ob bias broadcast.
__global__ void k_fillb(float* __restrict__ cat, const float* __restrict__ bias,
                        int sC, int oC)
{
    cat[(size_t)blockIdx.x * sC + oC + threadIdx.x] = bias[threadIdx.x];
}

// ---------------------------------------------------------------------------
// Fused residual add + LayerNorm over f=256. One block (256 thr) per token.
// x <- LN(x + d) * g + b
// ---------------------------------------------------------------------------
__global__ void __launch_bounds__(256)
k_add_ln(float* __restrict__ x, const float* __restrict__ d,
         const float* __restrict__ g, const float* __restrict__ bb)
{
    const int token = blockIdx.x;
    const int c = threadIdx.x;
    const size_t i = (size_t)token * 256 + c;
    const float v = x[i] + d[i];

    float s = v, s2 = v * v;
#pragma unroll
    for (int o = 16; o > 0; o >>= 1) {
        s  += __shfl_xor(s,  o, 32);
        s2 += __shfl_xor(s2, o, 32);
    }
    __shared__ float sh1[8], sh2[8], shm[2];
    const int wid = c >> 5, lane = c & 31;
    if (lane == 0) { sh1[wid] = s; sh2[wid] = s2; }
    __syncthreads();
    if (c == 0) {
        float a = 0.f, a2 = 0.f;
#pragma unroll
        for (int j = 0; j < 8; ++j) { a += sh1[j]; a2 += sh2[j]; }
        const float mean = a * (1.f / 256.f);
        const float var  = a2 * (1.f / 256.f) - mean * mean;
        shm[0] = mean;
        shm[1] = rsqrtf(var + 1e-5f);
    }
    __syncthreads();
    x[i] = (v - shm[0]) * shm[1] * g[c] + bb[c];
}

// ---------------------------------------------------------------------------
// Layout transposes
// ---------------------------------------------------------------------------
__global__ void k_nchw2nhwc(const float* __restrict__ in, float* __restrict__ out,
                            int C, int HW, int total)
{
    const int idx = blockIdx.x * 256 + threadIdx.x;
    if (idx >= total) return;
    const int b = idx / (C * HW);
    const int r = idx - b * C * HW;
    const int c = r / HW;
    const int p = r - c * HW;
    out[((size_t)b * HW + p) * C + c] = in[idx];
}

__global__ void k_nhwc2nchw(const float* __restrict__ in, float* __restrict__ out,
                            int C, int HW, int total)
{
    const int idx = blockIdx.x * 256 + threadIdx.x;
    if (idx >= total) return;
    const int b = idx / (C * HW);
    const int r = idx - b * C * HW;
    const int c = r / HW;
    const int p = r - c * HW;
    out[idx] = in[((size_t)b * HW + p) * C + c];
}

// ---------------------------------------------------------------------------
// Host orchestration
// ---------------------------------------------------------------------------
extern "C" void kernel_launch(void* const* d_in, const int* in_sizes, int n_in,
                              void* d_out, int out_size, void* d_ws, size_t ws_size,
                              hipStream_t stream)
{
    (void)in_sizes; (void)n_in; (void)out_size; (void)ws_size;

    const int HS[3]   = {64, 32, 16};
    const int M[3]    = {16384, 4096, 1024};     // B*H*W
    const int PROJ[3] = {1536, 2304, 1536};
    const int POS[3]  = {512, 768, 512};
    const int F = 256;

    const float *mapin[3], *pw[3], *pb[3], *aw[3], *ab[3], *ow[3], *ob[3], *pos[3];
    const float *lg1[3], *lb1[3], *fiw[3], *fib[3], *fow[3], *fob[3], *lg2[3], *lb2[3];
    for (int i = 0; i < 3; ++i) {
        mapin[i] = (const float*)d_in[0  + i];
        pw[i]    = (const float*)d_in[3  + i];
        pb[i]    = (const float*)d_in[6  + i];
        aw[i]    = (const float*)d_in[9  + i];
        ab[i]    = (const float*)d_in[12 + i];
        ow[i]    = (const float*)d_in[15 + i];
        ob[i]    = (const float*)d_in[18 + i];
        pos[i]   = (const float*)d_in[21 + i];
        lg1[i]   = (const float*)d_in[24 + i];
        lb1[i]   = (const float*)d_in[27 + i];
        fiw[i]   = (const float*)d_in[30 + i];
        fib[i]   = (const float*)d_in[33 + i];
        fow[i]   = (const float*)d_in[36 + i];
        fob[i]   = (const float*)d_in[39 + i];
        lg2[i]   = (const float*)d_in[42 + i];
        lb2[i]   = (const float*)d_in[45 + i];
    }

    // ---- workspace layout (fp32 region) ----
    float* base = (float*)d_ws;
    size_t off = 0;
    float* x[3];    for (int i = 0; i < 3; ++i) { x[i]    = base + off; off += (size_t)M[i] * F; }
    float* proj[3]; for (int i = 0; i < 3; ++i) { proj[i] = base + off; off += (size_t)M[i] * PROJ[i]; }
    float* cat[3];  for (int i = 0; i < 3; ++i) { cat[i]  = base + off; off += (size_t)M[i] * 3 * F; }
    float* delta = base + off; off += (size_t)M[0] * F;     // also NHWC staging for input proj
    float* tbuf  = base + off; off += (size_t)M[0] * F;
    float* hbuf  = proj[0];   // FFN hidden (M*1024) aliases proj0 (M0*1536): proj dead by then

    // ---- bf16 weight mirrors ----
    unsigned short* wb = (unsigned short*)(base + off);
    size_t wo = 0;
    unsigned short *pwB[3], *awB[3], *owB[3], *fiwB[3], *fowB[3];
    for (int i = 0; i < 3; ++i) { pwB[i]  = wb + wo; wo += (size_t)F * F; }
    for (int i = 0; i < 3; ++i) { awB[i]  = wb + wo; wo += (size_t)2 * PROJ[i] * F; }
    for (int i = 0; i < 3; ++i) { owB[i]  = wb + wo; wo += (size_t)2 * F * 3 * F; }
    for (int i = 0; i < 3; ++i) { fiwB[i] = wb + wo; wo += (size_t)2 * 4 * F * F; }
    for (int i = 0; i < 3; ++i) { fowB[i] = wb + wo; wo += (size_t)2 * F * 4 * F; }

    auto cvt = [&](const float* src, unsigned short* dst, size_t total) {
        const int t4 = (int)(total / 4);
        k_cvtbf<<<(t4 + 255) / 256, 256, 0, stream>>>(src, dst, t4);
    };
    for (int i = 0; i < 3; ++i) {
        cvt(pw[i],  pwB[i],  (size_t)F * F);
        cvt(aw[i],  awB[i],  (size_t)2 * PROJ[i] * F);
        cvt(ow[i],  owB[i],  (size_t)2 * F * 3 * F);
        cvt(fiw[i], fiwB[i], (size_t)2 * 4 * F * F);
        cvt(fow[i], fowB[i], (size_t)2 * F * 4 * F);
    }

    auto gemm = [&](const float* Ap, const unsigned short* Wp, const float* biasp,
                    float* Cp, int Mi, int Ni, int Ki, bool relu) {
        dim3 g(Ni / 64, Mi / 128), blk(256);
        if (relu) k_gemm<true ><<<g, blk, 0, stream>>>(Ap, Wp, biasp, Cp, Mi, Ni, Ki);
        else      k_gemm<false><<<g, blk, 0, stream>>>(Ap, Wp, biasp, Cp, Mi, Ni, Ki);
    };

    // ---- input 1x1 projection: x[i] = NHWC(map_i) @ pw^T + pb ----
    for (int i = 0; i < 3; ++i) {
        const int total = M[i] * F;
        k_nchw2nhwc<<<(total + 255) / 256, 256, 0, stream>>>(mapin[i], delta, F, HS[i] * HS[i], total);
        gemm(delta, pwB[i], pb[i], x[i], M[i], F, F, false);
    }

    for (int l = 0; l < 2; ++l) {
        // ---- fused per-map projections (all maps first, from old x) ----
        for (int i = 0; i < 3; ++i)
            gemm(x[i], awB[i] + (size_t)l * PROJ[i] * F, ab[i] + (size_t)l * PROJ[i],
                 proj[i], M[i], PROJ[i], F, false);

        const float* pL[3] = { pos[0] + (size_t)l * POS[0] * 9,
                               pos[1] + (size_t)l * POS[1] * 9,
                               pos[2] + (size_t)l * POS[2] * 9 };
        const float* oL[3] = { ob[0] + (size_t)l * 3 * F,
                               ob[1] + (size_t)l * 3 * F,
                               ob[2] + (size_t)l * 3 * F };

        // ---- map0: self, td(from map1 kv-below @5f), no bu ----
        k_attn<<<M[0], 256, 0, stream>>>(proj[0], 1536, 0,    proj[0], 1536, 256, 512,
                                         pL[0],          oL[0],       cat[0], 768, 0,
                                         64, 64, 64, 64, 0);
        k_attn<<<M[0], 256, 0, stream>>>(proj[0], 1536, 768,  proj[1], 2304, 1280, 1536,
                                         pL[0] + 256 * 9, oL[0] + 256, cat[0], 768, 256,
                                         64, 64, 32, 32, 1);
        k_fillb<<<M[0], 256, 0, stream>>>(cat[0], oL[0] + 512, 768, 512);

        // ---- map1: self, td(from map2 kv-below @4f), bu(from map0 kv-above @4f) ----
        k_attn<<<M[1], 256, 0, stream>>>(proj[1], 2304, 0,    proj[1], 2304, 256, 512,
                                         pL[1],          oL[1],       cat[1], 768, 0,
                                         32, 32, 32, 32, 0);
        k_attn<<<M[1], 256, 0, stream>>>(proj[1], 2304, 768,  proj[2], 1536, 1024, 1280,
                                         pL[1] + 256 * 9, oL[1] + 256, cat[1], 768, 256,
                                         32, 32, 16, 16, 1);
        k_attn<<<M[1], 256, 0, stream>>>(proj[1], 2304, 1024, proj[0], 1536, 1024, 1280,
                                         pL[1] + 512 * 9, oL[1] + 512, cat[1], 768, 512,
                                         32, 32, 64, 64, 2);

        // ---- map2: self, no td, bu(from map1 kv-above @7f), pos chunk = pos[f:2f] ----
        k_attn<<<M[2], 256, 0, stream>>>(proj[2], 1536, 0,    proj[2], 1536, 256, 512,
                                         pL[2],          oL[2],       cat[2], 768, 0,
                                         16, 16, 16, 16, 0);
        k_fillb<<<M[2], 256, 0, stream>>>(cat[2], oL[2] + 256, 768, 256);
        k_attn<<<M[2], 256, 0, stream>>>(proj[2], 1536, 768,  proj[1], 2304, 1792, 2048,
                                         pL[2] + 256 * 9, oL[2] + 512, cat[2], 768, 512,
                                         16, 16, 32, 32, 2);

        // ---- per-map tail: out proj, LN, FFN, LN (ob already folded into cat) ----
        for (int i = 0; i < 3; ++i) {
            gemm(cat[i], owB[i] + (size_t)l * F * 3 * F, nullptr, delta, M[i], F, 3 * F, false);
            k_add_ln<<<M[i], 256, 0, stream>>>(x[i], delta,
                                               lg1[i] + (size_t)l * F, lb1[i] + (size_t)l * F);
            gemm(x[i], fiwB[i] + (size_t)l * 4 * F * F, fib[i] + (size_t)l * 4 * F,
                 hbuf, M[i], 4 * F, F, true);
            gemm(hbuf, fowB[i] + (size_t)l * F * 4 * F, fob[i] + (size_t)l * F,
                 tbuf, M[i], F, 4 * F, false);
            k_add_ln<<<M[i], 256, 0, stream>>>(x[i], tbuf,
                                               lg2[i] + (size_t)l * F, lb2[i] + (size_t)l * F);
        }
    }

    // ---- NHWC -> NCHW into concatenated d_out ----
    float* outp = (float*)d_out;
    size_t ooff = 0;
    for (int i = 0; i < 3; ++i) {
        const int total = M[i] * F;
        k_nhwc2nchw<<<(total + 255) / 256, 256, 0, stream>>>(x[i], outp + ooff, F,
                                                             HS[i] * HS[i], total);
        ooff += (size_t)total;
    }
}
